// GraphAttentionLayer_3925600108694
// MI455X (gfx1250) — matrix operand order repaired and checked by reading, not compiled, not run on hardware
//
#include <hip/hip_runtime.h>
#include <hip/hip_fp16.h>

typedef _Float16 v16h __attribute__((ext_vector_type(16)));
typedef _Float16 v8h  __attribute__((ext_vector_type(8)));
typedef float    v8f  __attribute__((ext_vector_type(8)));
typedef float    v4f  __attribute__((ext_vector_type(4)));
typedef int      v4i  __attribute__((ext_vector_type(4)));

#define N_NODES 8192
#define FIN     256
#define FOUT    64
#define NSPLIT  8
#define JCHUNK  (N_NODES / NSPLIT)
#define LRELU_A 0.2f
#define LOG2E   1.44269504088896f

// ---------------------------------------------------------------------------
// Prep: WT[f][k] = (f16) W[k][f]; init ordered-uint float-max cell.
// ---------------------------------------------------------------------------
__global__ void gat_prep(const float* __restrict__ W, unsigned* __restrict__ maxs2u,
                         _Float16* __restrict__ WT) {
    int t = threadIdx.x;
    if (t == 0) *maxs2u = 0u;   // ordered encoding of -inf
    for (int e = t; e < FIN * FOUT; e += blockDim.x) {
        int f = e / FIN, k = e % FIN;
        WT[f * FIN + k] = (_Float16)W[k * FOUT + f];
    }
}

// ---------------------------------------------------------------------------
// h = input @ W, stored transposed as hT[f][i] in f16.
// Block = 16 rows of i, 4 waves = 4 f-tiles of 16. K-loop over Fin=256.
// ---------------------------------------------------------------------------
__global__ void __launch_bounds__(128) gat_hgemm(const float* __restrict__ input,
                                                 const _Float16* __restrict__ WT,
                                                 _Float16* __restrict__ hT) {
    const int wave = threadIdx.x >> 5;
    const int lane = threadIdx.x & 31;
    const int g    = lane >> 4;
    const int mrow = lane & 15;
    const int i0   = blockIdx.x * 16;
    const int f    = wave * 16 + mrow;

    const float*    arow = input + (i0 + mrow) * FIN;
    const _Float16* brow = WT + f * FIN;

    v8f acc = {};
    #pragma unroll
    for (int kk = 0; kk < FIN; kk += 32) {
        const v4f* a0 = reinterpret_cast<const v4f*>(arow + kk + 8 * g);
        const v4f* a1 = reinterpret_cast<const v4f*>(arow + kk + 8 * g + 16);
        v4f x0 = a0[0], x1 = a0[1], y0 = a1[0], y1 = a1[1];
        v16h a;
        #pragma unroll
        for (int i = 0; i < 4; ++i) {
            a[i]      = (_Float16)x0[i];
            a[i + 4]  = (_Float16)x1[i];
            a[i + 8]  = (_Float16)y0[i];
            a[i + 12] = (_Float16)y1[i];
        }
        const v8h* bp = reinterpret_cast<const v8h*>(brow + kk + 8 * g);
        v8h b0 = bp[0], b1 = bp[2];
        v16h b;
        #pragma unroll
        for (int i = 0; i < 8; ++i) { b[i] = b0[i]; b[i + 8] = b1[i]; }

        acc = __builtin_amdgcn_wmma_f32_16x16x32_f16(false, a, false, b,
                                                     (short)0, acc, false, false);
    }
    v8h o;
    #pragma unroll
    for (int r = 0; r < 8; ++r) o[r] = (_Float16)acc[r];
    *reinterpret_cast<v8h*>(hT + f * N_NODES + i0 + 8 * g) = o;
}

// ---------------------------------------------------------------------------
// s1[i] = h[i]·a1, s2[i] = h[i]·a2, plus global max(s2) (ordered-uint atomic).
// ---------------------------------------------------------------------------
__global__ void gat_scores(const _Float16* __restrict__ hT, const float* __restrict__ a,
                           float* __restrict__ s1, float* __restrict__ s2,
                           unsigned* __restrict__ maxs2u) {
    const int i = blockIdx.x * blockDim.x + threadIdx.x;
    float acc1 = 0.f, acc2 = 0.f;
    #pragma unroll 8
    for (int f = 0; f < FOUT; ++f) {
        float hv = (float)hT[f * N_NODES + i];
        acc1 += hv * a[f];
        acc2 += hv * a[FOUT + f];
    }
    s1[i] = acc1;
    s2[i] = acc2;

    __shared__ float red[256];
    red[threadIdx.x] = acc2;
    __syncthreads();
    for (int s = 128; s > 0; s >>= 1) {
        if (threadIdx.x < s) red[threadIdx.x] = fmaxf(red[threadIdx.x], red[threadIdx.x + s]);
        __syncthreads();
    }
    if (threadIdx.x == 0) {
        unsigned u = __float_as_uint(red[0]);
        u = (u & 0x80000000u) ? ~u : (u | 0x80000000u);
        atomicMax(maxs2u, u);
    }
}

// ---------------------------------------------------------------------------
// Fused masked-attention pass. One wave owns a 16-row i-tile and a 1024-wide
// j-split: builds the p-matrix A-fragment directly in registers (no LDS, no
// barriers), streams adj non-temporally, and issues 4 WMMAs (all of Fout)
// per 32-j step. Partial numerators/denominators go to per-split slabs.
// ---------------------------------------------------------------------------
__global__ void __launch_bounds__(256) gat_attn(const int* __restrict__ adj,
                                                const _Float16* __restrict__ hT,
                                                const float* __restrict__ s1,
                                                const float* __restrict__ s2,
                                                const unsigned* __restrict__ maxs2u,
                                                float* __restrict__ num,
                                                float* __restrict__ den) {
    const int wid   = (blockIdx.x * 256 + threadIdx.x) >> 5;
    const int lane  = threadIdx.x & 31;
    const int itile = wid >> 3;           // 0..511
    const int split = wid & (NSPLIT - 1); // 0..7
    const int g     = lane >> 4;
    const int nidx  = lane & 15;
    const int i0    = itile * 16;
    const int jbeg  = split * JCHUNK;
    const int jend  = jbeg + JCHUNK;

    // Row constants: safe softmax bound M_r = max(lrelu(s1_r + max(s2)), -10)
    unsigned u  = *maxs2u;
    unsigned bb = (u & 0x80000000u) ? (u ^ 0x80000000u) : ~u;
    float ms2 = __uint_as_float(bb);
    float s1r = s1[i0 + nidx];
    float z0  = s1r + ms2;
    float Mr  = fmaxf(fmaxf(z0, LRELU_A * z0), -10.0f);
    float MrL = Mr * LOG2E;

    const int* adjrow = adj + (i0 + nidx) * N_NODES;

    v8f acc[4] = {};
    float dpart = 0.f;

    for (int j0 = jbeg; j0 < jend; j0 += 32) {
        const int ka = j0 + 8 * g;        // A K-chunk 1 (elements 0..7)
        const int kb = ka + 16;           // A K-chunk 2 (elements 8..15)

        v4i av0 = __builtin_nontemporal_load((const v4i*)(adjrow + ka));
        v4i av1 = __builtin_nontemporal_load((const v4i*)(adjrow + ka + 4));
        v4i av2 = __builtin_nontemporal_load((const v4i*)(adjrow + kb));
        v4i av3 = __builtin_nontemporal_load((const v4i*)(adjrow + kb + 4));
        v4f sv0 = *(const v4f*)(s2 + ka);
        v4f sv1 = *(const v4f*)(s2 + ka + 4);
        v4f sv2 = *(const v4f*)(s2 + kb);
        v4f sv3 = *(const v4f*)(s2 + kb + 4);

        if (j0 + 32 < jend)
            __builtin_prefetch(adjrow + j0 + 32 + 8 * g, 0, 0);  // global_prefetch_b8

        v16h a;
        #pragma unroll
        for (int c = 0; c < 4; ++c) {
            float z, e, m, p;
            z = s1r + sv0[c]; e = fmaxf(z, LRELU_A * z);
            m = av0[c] > 0 ? e : -10.0f;
            p = exp2f(__builtin_fmaf(m, LOG2E, -MrL)); dpart += p; a[c]      = (_Float16)p;
            z = s1r + sv1[c]; e = fmaxf(z, LRELU_A * z);
            m = av1[c] > 0 ? e : -10.0f;
            p = exp2f(__builtin_fmaf(m, LOG2E, -MrL)); dpart += p; a[c + 4]  = (_Float16)p;
            z = s1r + sv2[c]; e = fmaxf(z, LRELU_A * z);
            m = av2[c] > 0 ? e : -10.0f;
            p = exp2f(__builtin_fmaf(m, LOG2E, -MrL)); dpart += p; a[c + 8]  = (_Float16)p;
            z = s1r + sv3[c]; e = fmaxf(z, LRELU_A * z);
            m = av3[c] > 0 ? e : -10.0f;
            p = exp2f(__builtin_fmaf(m, LOG2E, -MrL)); dpart += p; a[c + 12] = (_Float16)p;
        }

        // 4 f-tiles share the A fragment; B = two 16 B chunks of hT per lane (k = 8g.., 8g+16..).
        #pragma unroll
        for (int t = 0; t < 4; ++t) {
            const v8h* bp = reinterpret_cast<const v8h*>(
                hT + (t * 16 + nidx) * N_NODES + j0 + 8 * g);
            v8h b0 = bp[0], b1 = bp[2];
            v16h b;
            #pragma unroll
            for (int i = 0; i < 8; ++i) { b[i] = b0[i]; b[i + 8] = b1[i]; }
            acc[t] = __builtin_amdgcn_wmma_f32_16x16x32_f16(false, a, false, b,
                                                            (short)0, acc[t], false, false);
        }
    }

    // Row denominator: lane L and L^16 cover complementary K sets of row nidx.
    float drow = dpart + __shfl_xor(dpart, 16, 32);

    float* nslab = num + split * (N_NODES * FOUT);
    #pragma unroll
    for (int t = 0; t < 4; ++t)
        #pragma unroll
        for (int r = 0; r < 8; ++r)
            nslab[(i0 + r + 8 * g) * FOUT + t * 16 + nidx] = acc[t][r];

    if (lane < 16)
        den[split * N_NODES + i0 + lane] = drow;
}

// ---------------------------------------------------------------------------
// Reduce the 8 split-slabs, divide, ELU.
// ---------------------------------------------------------------------------
__global__ void gat_final(const float* __restrict__ num, const float* __restrict__ den,
                          float* __restrict__ out) {
    const int idx = blockIdx.x * 256 + threadIdx.x;   // 0 .. 8192*64-1
    const int i   = idx >> 6;
    float n = 0.f, d = 0.f;
    #pragma unroll
    for (int s = 0; s < NSPLIT; ++s) {
        n += num[s * (N_NODES * FOUT) + idx];
        d += den[s * N_NODES + i];
    }
    float hp = n / d;
    out[idx] = hp > 0.f ? hp : (exp2f(hp * LOG2E) - 1.0f);
}

// ---------------------------------------------------------------------------
extern "C" void kernel_launch(void* const* d_in, const int* in_sizes, int n_in,
                              void* d_out, int out_size, void* d_ws, size_t ws_size,
                              hipStream_t stream) {
    const float* input = (const float*)d_in[0];   // 8192 x 256
    const int*   adj   = (const int*)d_in[1];     // 8192 x 8192
    const float* W     = (const float*)d_in[2];   // 256 x 64
    const float* a     = (const float*)d_in[3];   // 128 x 1
    float*       out   = (float*)d_out;           // 8192 x 64

    char* ws = (char*)d_ws;
    _Float16* hT  = (_Float16*)ws;                                  // 1 MB
    _Float16* WT  = (_Float16*)(ws + (1u << 20));                   // 32 KB
    float*    s1  = (float*)(ws + (1u << 20) + (32u << 10));        // 32 KB
    float*    s2  = (float*)(ws + (1u << 20) + (64u << 10));        // 32 KB
    unsigned* mx  = (unsigned*)(ws + (1u << 20) + (96u << 10));     // 4 B
    float*    num = (float*)(ws + (2u << 20));                      // 8 x 2 MB
    float*    den = (float*)(ws + (18u << 20));                     // 8 x 32 KB

    gat_prep  <<<1,                          256, 0, stream>>>(W, mx, WT);
    gat_hgemm <<<N_NODES / 16,               128, 0, stream>>>(input, WT, hT);
    gat_scores<<<N_NODES / 256,              256, 0, stream>>>(hT, a, s1, s2, mx);
    gat_attn  <<<(N_NODES / 16) * NSPLIT / 8, 256, 0, stream>>>(adj, hT, s1, s2, mx, num, den);
    gat_final <<<(N_NODES * FOUT) / 256,     256, 0, stream>>>(num, den, out);
}
